// Conv1dSubsamplingShrink_63866163692257
// MI455X (gfx1250) — compile-verified, matrix operand order checked by reading, not run on hardware
//
#include <hip/hip_runtime.h>
#include <hip/hip_bf16.h>
#include <stdint.h>

typedef _Float16 v16h __attribute__((ext_vector_type(16)));
typedef _Float16 v8h  __attribute__((ext_vector_type(8)));
typedef _Float16 v2h  __attribute__((ext_vector_type(2)));
typedef float    v8f  __attribute__((ext_vector_type(8)));
typedef float    v2f  __attribute__((ext_vector_type(2)));

// ---------------- constants ----------------
#define T_IN   3000
#define BSZ    32
#define D_IN   80
#define T1     1500      // after conv1 stride 2
#define T2     750       // after conv2 stride 2
#define CH     512       // post-GLU channels
#define KW     5
#define PADW   2
#define KDIM1  400       // 80*5
#define KT1    13        // ceil(400/32)
#define KDIM2  2560      // 512*5
#define KT2    80        // 2560/32 exact
#define NG1    188       // ceil(T1/8)  (one t-step = 32 columns = 2 N-tiles per wave)
#define NG2    94        // ceil(T2/8)

// K index within a 16-bit WMMA tile for (vgpr j, lane-group g): pairs (kb, kb+1)
__device__ __forceinline__ int kbase(int j, int g) {
    return ((j < 4) ? 2 * j : 8 + 2 * j) + 8 * g;
}

__device__ __forceinline__ float sigf(float x) {
    return 1.0f / (1.0f + __expf(-x));
}

// ---------------- weight packing ----------------
// packed layout: [mtile][ktile][lane][8 dwords]; one v16h per lane per tile.
__global__ void pack_w1(const float* __restrict__ w1, uint32_t* __restrict__ pw) {
    int tid = blockIdx.x * blockDim.x + threadIdx.x;
    if (tid >= 64 * KT1 * 32) return;
    int lane = tid & 31;
    int kt   = (tid >> 5) % KT1;
    int mt   = tid / (KT1 * 32);
    int g    = lane >> 4;
    int co   = mt * 16 + (lane & 15);
#pragma unroll
    for (int j = 0; j < 8; ++j) {
        int kk = kt * 32 + kbase(j, g);     // even
        _Float16 h0 = (_Float16)0.0f, h1 = (_Float16)0.0f;
        if (kk < KDIM1) {                   // kk = k*80 + cin (cin even, pair same k)
            int k = kk / 80, cin = kk - k * 80;
            h0 = (_Float16)w1[co * KDIM1 + cin * KW + k];
            h1 = (_Float16)w1[co * KDIM1 + (cin + 1) * KW + k];
        }
        union { uint32_t u; _Float16 h[2]; } q;
        q.h[0] = h0; q.h[1] = h1;
        pw[tid * 8 + j] = q.u;
    }
}

__global__ void pack_w2(const float* __restrict__ w2, uint32_t* __restrict__ pw) {
    int tid = blockIdx.x * blockDim.x + threadIdx.x;
    if (tid >= 64 * KT2 * 32) return;
    int lane = tid & 31;
    int kt   = (tid >> 5) % KT2;
    int mt   = tid / (KT2 * 32);
    int g    = lane >> 4;
    int co   = mt * 16 + (lane & 15);
#pragma unroll
    for (int j = 0; j < 8; ++j) {
        int kk = kt * 32 + kbase(j, g);     // kk = k*512 + c, always < 2560
        int k = kk >> 9, c = kk & 511;
        union { uint32_t u; _Float16 h[2]; } q;
        q.h[0] = (_Float16)w2[co * KDIM2 + c * KW + k];
        q.h[1] = (_Float16)w2[co * KDIM2 + (c + 1) * KW + k];
        pw[tid * 8 + j] = q.u;
    }
}

// ---------------- layer 1: GEMM (im2col from src) + GLU -> h (f16) ----------------
// grid = (NG1, 8): blockIdx.y = m-group (4 value + 4 gate tiles, LDS-staged, shared by
// all 8 waves); each wave owns one t1 step = 32 columns (2 N-tiles, batches 0-15/16-31).
__global__ void __launch_bounds__(256, 1)
gemm1_glu(const float* __restrict__ src,   // (T,B,D) f32
          const float* __restrict__ bias1, // (1024,)
          const uint32_t* __restrict__ pw1,
          _Float16* __restrict__ h)        // (T1,B,512) f16
{
    __shared__ uint32_t sA[8 * 32 * 8];      // 8 tiles x 32 lanes x 8 dwords = 8KB

    const int lane = threadIdx.x & 31;
    const int wid  = threadIdx.x >> 5;
    const int mgy  = blockIdx.y;             // 0..7
    const int g    = lane >> 4;
    const int nl   = lane & 15;

    const int u      = blockIdx.x * 8 + wid; // t1 index
    const bool actv  = (u < T1);
    const int t1     = actv ? u : (T1 - 1);

    const v8f zero = {0.f, 0.f, 0.f, 0.f, 0.f, 0.f, 0.f, 0.f};
    v8f accv[2][4], accg[2][4];
#pragma unroll
    for (int nw = 0; nw < 2; ++nw)
#pragma unroll
        for (int i = 0; i < 4; ++i) { accv[nw][i] = zero; accg[nw][i] = zero; }

    for (int kt = 0; kt < KT1; ++kt) {
        __syncthreads();
        {   // cooperative A-tile stage: 8 tiles (4 value + 4 gate) for this m-group
            int t    = threadIdx.x;
            int tile = t >> 5, ln = t & 31;
            int mt   = (tile < 4) ? (mgy * 4 + tile) : (32 + mgy * 4 + (tile - 4));
            const uint4* s = (const uint4*)(pw1 + ((size_t)(mt * KT1 + kt) * 32 + ln) * 8);
            uint4* d = (uint4*)&sA[t * 8];
            d[0] = s[0];
            d[1] = s[1];
        }
        __syncthreads();

        v16h bv[2];
#pragma unroll
        for (int nw = 0; nw < 2; ++nw) {
            const int b = nw * 16 + nl;      // batch
#pragma unroll
            for (int j = 0; j < 8; ++j) {
                int kk = kt * 32 + kbase(j, g);   // even; pair (kk,kk+1) same k
                v2f x = {0.f, 0.f};
                if (kk < KDIM1) {
                    int k = kk / 80, cin = kk - k * 80;
                    int tin = 2 * t1 + k - PADW;
                    if (tin >= 0 && tin < T_IN)
                        x = *(const v2f*)(src + ((size_t)tin * BSZ + b) * D_IN + cin);
                }
                bv[nw][2 * j]     = (_Float16)x[0];
                bv[nw][2 * j + 1] = (_Float16)x[1];
            }
        }
#pragma unroll
        for (int i = 0; i < 4; ++i) {
            v16h av = *(const v16h*)&sA[(i * 32 + lane) * 8];
#pragma unroll
            for (int nw = 0; nw < 2; ++nw)
                accv[nw][i] = __builtin_amdgcn_wmma_f32_16x16x32_f16(
                    false, av, false, bv[nw], (short)0, accv[nw][i], false, false);
            v16h ag = *(const v16h*)&sA[((4 + i) * 32 + lane) * 8];
#pragma unroll
            for (int nw = 0; nw < 2; ++nw)
                accg[nw][i] = __builtin_amdgcn_wmma_f32_16x16x32_f16(
                    false, ag, false, bv[nw], (short)0, accg[nw][i], false, false);
        }
    }

    if (actv) {
#pragma unroll
        for (int nw = 0; nw < 2; ++nw) {
#pragma unroll
            for (int i = 0; i < 4; ++i) {
                int cbase = (mgy * 4 + i) * 16 + 8 * g;   // post-GLU channel base
                v8h hv;
#pragma unroll
                for (int r = 0; r < 8; ++r) {
                    float v  = accv[nw][i][r] + bias1[cbase + r];
                    float gg = accg[nw][i][r] + bias1[CH + cbase + r];
                    hv[r] = (_Float16)(v * sigf(gg));
                }
                size_t n = (size_t)t1 * 32 + nw * 16 + nl;   // (t1, b)
                *(v8h*)(h + n * CH + cbase) = hv;
            }
        }
    }
}

// ---------------- layer 2: GEMM (im2col from h) + GLU -> out (f32, (T2,B,512)) ------
__global__ void __launch_bounds__(256, 1)
gemm2_glu(const _Float16* __restrict__ h,  // (T1,B,512) f16
          const float* __restrict__ bias2, // (1024,)
          const uint32_t* __restrict__ pw2,
          float* __restrict__ out)         // (T2,B,512) f32
{
    __shared__ uint32_t sA[8 * 32 * 8];      // 8KB

    const int lane = threadIdx.x & 31;
    const int wid  = threadIdx.x >> 5;
    const int mgy  = blockIdx.y;
    const int g    = lane >> 4;
    const int nl   = lane & 15;

    const int u     = blockIdx.x * 8 + wid;  // t2 index
    const bool actv = (u < T2);
    const int t2    = actv ? u : (T2 - 1);

    const v8f zero = {0.f, 0.f, 0.f, 0.f, 0.f, 0.f, 0.f, 0.f};
    v8f accv[2][4], accg[2][4];
#pragma unroll
    for (int nw = 0; nw < 2; ++nw)
#pragma unroll
        for (int i = 0; i < 4; ++i) { accv[nw][i] = zero; accg[nw][i] = zero; }

    for (int kt = 0; kt < KT2; ++kt) {
        __syncthreads();
        {
            int t    = threadIdx.x;
            int tile = t >> 5, ln = t & 31;
            int mt   = (tile < 4) ? (mgy * 4 + tile) : (32 + mgy * 4 + (tile - 4));
            const uint4* s = (const uint4*)(pw2 + ((size_t)(mt * KT2 + kt) * 32 + ln) * 8);
            uint4* d = (uint4*)&sA[t * 8];
            d[0] = s[0];
            d[1] = s[1];
        }
        __syncthreads();

        v16h bv[2];
#pragma unroll
        for (int nw = 0; nw < 2; ++nw) {
            const int b = nw * 16 + nl;
#pragma unroll
            for (int j = 0; j < 8; ++j) {
                int kk = kt * 32 + kbase(j, g);   // kk = k*512 + c, c even
                int k = kk >> 9, c = kk & 511;
                int tin = 2 * t2 + k - PADW;
                v2h x = {(_Float16)0.0f, (_Float16)0.0f};
                if (tin >= 0 && tin < T1)
                    x = *(const v2h*)(h + ((size_t)tin * BSZ + b) * CH + c);
                bv[nw][2 * j]     = x[0];
                bv[nw][2 * j + 1] = x[1];
            }
        }
#pragma unroll
        for (int i = 0; i < 4; ++i) {
            v16h av = *(const v16h*)&sA[(i * 32 + lane) * 8];
#pragma unroll
            for (int nw = 0; nw < 2; ++nw)
                accv[nw][i] = __builtin_amdgcn_wmma_f32_16x16x32_f16(
                    false, av, false, bv[nw], (short)0, accv[nw][i], false, false);
            v16h ag = *(const v16h*)&sA[((4 + i) * 32 + lane) * 8];
#pragma unroll
            for (int nw = 0; nw < 2; ++nw)
                accg[nw][i] = __builtin_amdgcn_wmma_f32_16x16x32_f16(
                    false, ag, false, bv[nw], (short)0, accg[nw][i], false, false);
        }
    }

    if (actv) {
#pragma unroll
        for (int nw = 0; nw < 2; ++nw) {
#pragma unroll
            for (int i = 0; i < 4; ++i) {
                int cbase = (mgy * 4 + i) * 16 + 8 * g;
                v8f ov;
#pragma unroll
                for (int r = 0; r < 8; ++r) {
                    float v  = accv[nw][i][r] + bias2[cbase + r];
                    float gg = accg[nw][i][r] + bias2[CH + cbase + r];
                    ov[r] = v * sigf(gg);
                }
                size_t n = (size_t)t2 * 32 + nw * 16 + nl;   // (t2, b)
                *(v8f*)(out + n * CH + cbase) = ov;
            }
        }
    }
}

// ---------------- output lengths ----------------
__global__ void out_lens(const int* __restrict__ sl, float* __restrict__ o) {
    int i = threadIdx.x;
    if (i < BSZ) {
        int L = sl[i];
        L = (L + 1) >> 1;   // floor((L-1)/2 + 1) for stride 2
        L = (L + 1) >> 1;
        o[i] = (float)L;
    }
}

extern "C" void kernel_launch(void* const* d_in, const int* in_sizes, int n_in,
                              void* d_out, int out_size, void* d_ws, size_t ws_size,
                              hipStream_t stream) {
    (void)in_sizes; (void)n_in; (void)ws_size;
    const float* src = (const float*)d_in[0];
    const int*   sl  = (const int*)d_in[1];
    const float* w1  = (const float*)d_in[2];
    const float* b1  = (const float*)d_in[3];
    const float* w2  = (const float*)d_in[4];
    const float* b2  = (const float*)d_in[5];
    float* out = (float*)d_out;

    // workspace carve-up
    uint8_t* ws = (uint8_t*)d_ws;
    uint32_t* pw1 = (uint32_t*)(ws);                       // 64*13*32*8 dw  = 0.85 MB
    uint32_t* pw2 = (uint32_t*)(ws + (1u << 20));          // 64*80*32*8 dw  = 5.24 MB
    _Float16* hbuf = (_Float16*)(ws + (6u << 20));         // 1500*32*512 h  = 49.2 MB

    pack_w1<<<(64 * KT1 * 32 + 255) / 256, 256, 0, stream>>>(w1, pw1);
    pack_w2<<<(64 * KT2 * 32 + 255) / 256, 256, 0, stream>>>(w2, pw2);
    gemm1_glu<<<dim3(NG1, 8), 256, 0, stream>>>(src, b1, pw1, hbuf);
    gemm2_glu<<<dim3(NG2, 8), 256, 0, stream>>>(hbuf, b2, pw2, out);
    out_lens<<<1, 32, 0, stream>>>(sl, out + (out_size - BSZ));
}
